// QuantizedLinear_38062000177824
// MI455X (gfx1250) — 8-bit matrix operand order repaired and checked by reading, compiled, not run on hardware
//
#include <hip/hip_runtime.h>


typedef int v8i __attribute__((ext_vector_type(8)));
typedef int v4i __attribute__((ext_vector_type(4)));

typedef __attribute__((address_space(1))) v4i as1_v4i;
typedef __attribute__((address_space(3))) v4i as3_v4i;

#define M_DIM 8192
#define N_DIM 4096
#define K_DIM 4096
#define BM 128
#define BN 128
#define BK 128
#define BKP 144

#if __has_builtin(__builtin_amdgcn_global_load_async_to_lds_b128)
#define ASYNC_LDS 1
#else
#define ASYNC_LDS 0
#endif

__device__ __forceinline__ void cp_b128_to_lds(const unsigned char* g, unsigned char* l) {
#if ASYNC_LDS
  __builtin_amdgcn_global_load_async_to_lds_b128((as1_v4i*)g, (as3_v4i*)l, 0, 0);
#else
  *(v4i*)l = *(const v4i*)g;
#endif
}

__device__ __forceinline__ void wait_stage(bool next_issued) {
#if ASYNC_LDS
  if (next_issued) {
#if __has_builtin(__builtin_amdgcn_s_wait_asynccnt)
    __builtin_amdgcn_s_wait_asynccnt(8);
#else
    asm volatile("s_wait_asynccnt 8" ::: "memory");
#endif
  } else {
#if __has_builtin(__builtin_amdgcn_s_wait_asynccnt)
    __builtin_amdgcn_s_wait_asynccnt(0);
#else
    asm volatile("s_wait_asynccnt 0" ::: "memory");
#endif
  }
#endif
  __syncthreads();
}

__global__ __launch_bounds__(256) void pack_i32_to_i8(const int* __restrict__ in,
                                                      unsigned int* __restrict__ out,
                                                      int n4) {
  int i = blockIdx.x * 256 + threadIdx.x;
  if (i < n4) {
    v4i v = *(const v4i*)(in + 4 * (size_t)i);
    unsigned int p = ((unsigned)v.x & 255u) | (((unsigned)v.y & 255u) << 8) |
                     (((unsigned)v.z & 255u) << 16) | (((unsigned)v.w & 255u) << 24);
    out[i] = p;
  }
}

__global__ __launch_bounds__(256, 1) void qlinear_i8_wmma(
    const unsigned char* __restrict__ x8, const unsigned char* __restrict__ w8,
    const int* __restrict__ bias, const float* __restrict__ wscale,
    const float* __restrict__ in_scale_p, const float* __restrict__ out_scale_p,
    const int* __restrict__ zp_p, float* __restrict__ out) {
  __shared__ unsigned char sA[2][BM * BKP];
  __shared__ unsigned char sB[2][BN * BKP];

  const int t = threadIdx.x;
  const int lane = t & 31;
  const int laneHalf = lane >> 4;
  const int laneMod = lane & 15;
  const int wave = t >> 5;
  const int waveM = (wave >> 2) * 64;
  const int waveN = (wave & 3) * 32;
  const int m0 = blockIdx.y * BM;
  const int n0 = blockIdx.x * BN;

  auto stage = [&](int buf, int k0) {
#pragma unroll
    for (int i = 0; i < 4; ++i) {
      int lin = (t + i * 256) << 4;
      int row = lin >> 7;
      int kb = lin & 127;
      cp_b128_to_lds(x8 + (size_t)(m0 + row) * K_DIM + (size_t)(k0 + kb),
                     &sA[buf][row * BKP + kb]);
    }
#pragma unroll
    for (int i = 0; i < 4; ++i) {
      int lin = (t + i * 256) << 4;
      int row = lin >> 7;
      int kb = lin & 127;
      cp_b128_to_lds(w8 + (size_t)(n0 + row) * K_DIM + (size_t)(k0 + kb),
                     &sB[buf][row * BKP + kb]);
    }
  };

  v8i acc[4][2];
#pragma unroll
  for (int m = 0; m < 4; ++m)
#pragma unroll
    for (int n = 0; n < 2; ++n) acc[m][n] = (v8i){0, 0, 0, 0, 0, 0, 0, 0};

  stage(0, 0);
  int buf = 0;
  for (int k0 = 0; k0 < K_DIM; k0 += BK) {
    const bool has_next = (k0 + BK) < K_DIM;
    if (has_next) stage(buf ^ 1, k0 + BK);
    wait_stage(has_next);

    const unsigned char* At = sA[buf];
    const unsigned char* Bt = sB[buf];
#pragma unroll
    for (int kk = 0; kk < 2; ++kk) {
      const int ks = kk << 6;
      v8i bfrag[2];
#pragma unroll
      for (int n = 0; n < 2; ++n) {
        const unsigned char* p =
            Bt + (waveN + n * 16 + laneMod) * BKP + ks + (laneHalf << 4);
        union { v4i q[2]; v8i v; } ub;
        ub.q[0] = *(const v4i*)(p);
        ub.q[1] = *(const v4i*)(p + 32);
        bfrag[n] = ub.v;
      }
#pragma unroll
      for (int m = 0; m < 4; ++m) {
        const unsigned char* p =
            At + (waveM + m * 16 + laneMod) * BKP + ks + (laneHalf << 4);
        union { unsigned long long d[4]; v8i v; } ua;
        ua.d[0] = *(const unsigned long long*)(p);
        ua.d[1] = *(const unsigned long long*)(p + 8);
        ua.d[2] = *(const unsigned long long*)(p + 32);
        ua.d[3] = *(const unsigned long long*)(p + 40);
#pragma unroll
        for (int n = 0; n < 2; ++n) {
          acc[m][n] = __builtin_amdgcn_wmma_i32_16x16x64_iu8(
               true, ua.v,  true, bfrag[n], acc[m][n],
               false,  false);
        }
      }
    }
    __syncthreads();
    buf ^= 1;
  }

  const float is = in_scale_p[0];
  const float os_inv = 1.0f / out_scale_p[0];
  const float zpf = (float)zp_p[0];
#pragma unroll
  for (int n = 0; n < 2; ++n) {
    const int col = n0 + waveN + n * 16 + laneMod;
    const float scl = is * wscale[col] * os_inv;
    const int bv = bias[col];
#pragma unroll
    for (int m = 0; m < 4; ++m) {
      const int rowbase = m0 + waveM + m * 16 + (laneHalf << 3);
#pragma unroll
      for (int i = 0; i < 8; ++i) {
        float f = (float)(acc[m][n][i] + bv) * scl + zpf;
        f = __builtin_rintf(f);
        f = fminf(fmaxf(f, -128.0f), 127.0f);
        out[(size_t)(rowbase + i) * N_DIM + col] = f;
      }
    }
  }
}

__global__ __launch_bounds__(256) void qlinear_naive(
    const int* __restrict__ x, const int* __restrict__ w,
    const int* __restrict__ bias, const float* __restrict__ wscale,
    const float* __restrict__ in_scale_p, const float* __restrict__ out_scale_p,
    const int* __restrict__ zp_p, float* __restrict__ out) {
  size_t idx = (size_t)blockIdx.x * 256 + threadIdx.x;
  if (idx >= (size_t)M_DIM * N_DIM) return;
  int row = (int)(idx / N_DIM), col = (int)(idx % N_DIM);
  int acc = bias[col];
  const int* xr = x + (size_t)row * K_DIM;
  const int* wr = w + (size_t)col * K_DIM;
  for (int k = 0; k < K_DIM; ++k) acc += xr[k] * wr[k];
  float f = (float)acc * (in_scale_p[0] * wscale[col] / out_scale_p[0]) + (float)zp_p[0];
  f = __builtin_rintf(f);
  f = fminf(fmaxf(f, -128.0f), 127.0f);
  out[idx] = f;
}

extern "C" void kernel_launch(void* const* d_in, const int* in_sizes, int n_in,
                              void* d_out, int out_size, void* d_ws, size_t ws_size,
                              hipStream_t stream) {
  const int* x = (const int*)d_in[0];
  const int* w = (const int*)d_in[1];
  const int* bias = (const int*)d_in[2];
  const float* wscale = (const float*)d_in[3];
  const float* in_scale = (const float*)d_in[4];
  const float* out_scale = (const float*)d_in[5];
  const int* zp = (const int*)d_in[6];
  float* out = (float*)d_out;

  const size_t xBytes = (size_t)M_DIM * K_DIM;
  const size_t wBytes = (size_t)N_DIM * K_DIM;

  if (ws_size >= xBytes + wBytes) {
    unsigned char* x8 = (unsigned char*)d_ws;
    unsigned char* w8 = x8 + xBytes;

    const int xn4 = (int)(xBytes / 4);
    const int wn4 = (int)(wBytes / 4);
    pack_i32_to_i8<<<(xn4 + 255) / 256, 256, 0, stream>>>(x, (unsigned int*)x8, xn4);
    pack_i32_to_i8<<<(wn4 + 255) / 256, 256, 0, stream>>>(w, (unsigned int*)w8, wn4);

    dim3 grid(N_DIM / BN, M_DIM / BM);
    qlinear_i8_wmma<<<grid, 256, 0, stream>>>(x8, w8, bias, wscale, in_scale,
                                              out_scale, zp, out);
  } else {
    size_t total = (size_t)M_DIM * N_DIM;
    qlinear_naive<<<(unsigned)((total + 255) / 256), 256, 0, stream>>>(
        x, w, bias, wscale, in_scale, out_scale, zp, out);
  }
}
